// DTM_1408749273327
// MI455X (gfx1250) — matrix operand order repaired and checked by reading, not compiled, not run on hardware
//
#include <hip/hip_runtime.h>

typedef __attribute__((ext_vector_type(16))) __bf16 v16bf;
typedef __attribute__((ext_vector_type(8)))  __bf16 v8bf;
typedef __attribute__((ext_vector_type(8)))  float  v8f;

#define B_SZ   2048
#define H_SZ   512
#define K_TOP  50
#define T_SZ   10
#define V_SZ   50000
#define D_SZ   1024
#define R_SZ   (T_SZ * K_TOP)     // 500 rows of the logits matrix
#define EPSF   1e-10f

#if __has_builtin(__builtin_amdgcn_tensor_load_to_lds)
#define HAVE_TDM 1
#else
#define HAVE_TDM 0
#endif

// ---------------------------------------------------------------------------
// helpers
// ---------------------------------------------------------------------------
__device__ __forceinline__ v16bf pack16(v8bf lo, v8bf hi) {
  v16bf r;
#pragma unroll
  for (int i = 0; i < 8; ++i) { r[i] = lo[i]; r[i + 8] = hi[i]; }
  return r;
}

__device__ __forceinline__ float block_reduce_sum256(float v, float* buf) {
  int tid = threadIdx.x;
  buf[tid] = v; __syncthreads();
#pragma unroll
  for (int o = 128; o > 0; o >>= 1) {
    if (tid < o) buf[tid] += buf[tid + o];
    __syncthreads();
  }
  float r = buf[0]; __syncthreads();
  return r;
}

__device__ __forceinline__ void wait_tensorcnt0() {
#if __has_builtin(__builtin_amdgcn_s_wait_tensorcnt)
  __builtin_amdgcn_s_wait_tensorcnt(0);
#else
  asm volatile("s_wait_tensorcnt 0x0" ::: "memory");
#endif
}

#if HAVE_TDM
typedef __attribute__((ext_vector_type(4))) unsigned int u32x4;
typedef __attribute__((ext_vector_type(8))) int          i32x8;
typedef __attribute__((ext_vector_type(4))) int          i32x4;

// 1-D contiguous TDM copy: nelem f32 from global 'ga' -> LDS byte addr 'lds_addr'.
// D# per CDNA5 ISA ch.8: g0 = {count=1, lds_addr, global_addr[56:0], type=2},
// g1 = {data_size=4B, tensor_dim0=nelem, tensor_dim1=1, tile_dim0=nelem,
//       workgroup_mask=0 (not in cluster)}, groups 2/3 zero (<=2-D tensor).
__device__ __forceinline__ void tdm_1d_load(unsigned lds_addr,
                                            unsigned long long ga, int nelem) {
  u32x4 g0;
  g0.x = 1u;                                                 // count=1, user D#
  g0.y = lds_addr;                                           // bits 63:32
  g0.z = (unsigned)(ga & 0xFFFFFFFFull);                     // global_addr lo
  g0.w = (unsigned)((ga >> 32) & 0x01FFFFFFull) | 0x80000000u; // ga hi | type=2
  i32x8 g1;
  g1.s0 = (2 << 16);                                         // data_size = 4B
  g1.s1 = (nelem & 0xFFFF) << 16;                            // tensor_dim0 lo16
  g1.s2 = (int)(((unsigned)nelem >> 16) | (1u << 16));       // td0 hi | td1 lo=1
  g1.s3 = (nelem & 0xFFFF) << 16;                            // td1 hi=0 | tile_dim0
  g1.s4 = 0;                                                 // tile_dim1/2 unused
  g1.s5 = nelem;                                             // dim0 stride (unused 1-D)
  g1.s6 = 0;
  g1.s7 = 0;
  i32x4 z4 = {0, 0, 0, 0};
#if __clang_major__ >= 23
  i32x8 z8 = {0, 0, 0, 0, 0, 0, 0, 0};
  __builtin_amdgcn_tensor_load_to_lds(g0, g1, z4, z4, z8, 0);
#else
  __builtin_amdgcn_tensor_load_to_lds(g0, g1, z4, z4, 0);
#endif
}
#endif  // HAVE_TDM

// ---------------------------------------------------------------------------
// Generic WMMA GEMM:  C(MxN) = A(MxK) * B  (+bias, +relu)
//   BT_NK = false : B is (K x N) row-major   (encoder weights)
//   BT_NK = true  : B is (N x K) row-major   (word embeddings, contraction on K)
// Block tile 128x128, K-step 32, 8 waves (4x2), wave tile 32x64 (2x4 frags).
// f32 -> bf16 conversion happens while staging to LDS.
// ---------------------------------------------------------------------------
template<bool BT_NK, bool RELU, bool HASBIAS>
__global__ __launch_bounds__(256) void gemm_bf16_wmma(
    const float* __restrict__ A, const float* __restrict__ Bm,
    const float* __restrict__ bias, float* __restrict__ C,
    int M, int N, int K)
{
  __shared__ __bf16 sA[128][40];   // [m][k], pad->40 halves (80B rows, 16B aligned)
  __shared__ __bf16 sB[128][40];   // [n][k]

  const int tid  = threadIdx.x;
  const int lane = tid & 31;
  const int wave = tid >> 5;
  const int wm   = wave & 3;            // M sub-tile 0..3  (32 rows each)
  const int wn   = wave >> 2;           // N sub-tile 0..1  (64 cols each)
  const int m0   = blockIdx.y * 128;
  const int n0   = blockIdx.x * 128;

  v8f acc[2][4];
#pragma unroll
  for (int i = 0; i < 2; ++i)
#pragma unroll
    for (int j = 0; j < 4; ++j)
#pragma unroll
      for (int e = 0; e < 8; ++e) acc[i][j][e] = 0.f;

  for (int kt = 0; kt < K; kt += 32) {
    // ---- stage A tile 128x32 (f32 -> bf16) ----
#pragma unroll
    for (int it = 0; it < 4; ++it) {
      int idx = tid + it * 256;
      int row = idx >> 3;            // 0..127
      int c4  = (idx & 7) << 2;      // 0,4,...,28
      float4 v = make_float4(0.f, 0.f, 0.f, 0.f);
      if (m0 + row < M)
        v = *(const float4*)(A + (size_t)(m0 + row) * K + kt + c4);
      sA[row][c4 + 0] = (__bf16)v.x;
      sA[row][c4 + 1] = (__bf16)v.y;
      sA[row][c4 + 2] = (__bf16)v.z;
      sA[row][c4 + 3] = (__bf16)v.w;
    }
    // ---- stage B tile as [n][k] ----
    if constexpr (BT_NK) {           // B is N x K : direct rows
#pragma unroll
      for (int it = 0; it < 4; ++it) {
        int idx = tid + it * 256;
        int row = idx >> 3;          // n 0..127
        int c4  = (idx & 7) << 2;    // k
        float4 v = make_float4(0.f, 0.f, 0.f, 0.f);
        if (n0 + row < N)
          v = *(const float4*)(Bm + (size_t)(n0 + row) * K + kt + c4);
        sB[row][c4 + 0] = (__bf16)v.x;
        sB[row][c4 + 1] = (__bf16)v.y;
        sB[row][c4 + 2] = (__bf16)v.z;
        sB[row][c4 + 3] = (__bf16)v.w;
      }
    } else {                         // B is K x N : transpose while staging
#pragma unroll
      for (int it = 0; it < 4; ++it) {
        int idx = tid + it * 256;
        int kk = idx >> 5;           // 0..31
        int n4 = (idx & 31) << 2;    // 0..124
        float4 v = make_float4(0.f, 0.f, 0.f, 0.f);
        if (n0 + n4 + 3 < N)         // N is a multiple of 4 on this path
          v = *(const float4*)(Bm + (size_t)(kt + kk) * N + n0 + n4);
        sB[n4 + 0][kk] = (__bf16)v.x;
        sB[n4 + 1][kk] = (__bf16)v.y;
        sB[n4 + 2][kk] = (__bf16)v.z;
        sB[n4 + 3][kk] = (__bf16)v.w;
      }
    }
    __syncthreads();

    // ---- fragment loads per ISA VGPR layouts ----
    const int arow = wm * 32 + (lane & 15);
    const int ak0  = (lane < 16) ? 0 : 8;      // A: two 8-half runs at k0, k0+16
    const int bcol = wn * 64 + (lane & 15);
    const int bk0  = (lane < 16) ? 0 : 8;      // B: two 8-half runs at k0, k0+16

    v16bf af[2], bfr[4];
#pragma unroll
    for (int fm = 0; fm < 2; ++fm) {
      const __bf16* p = &sA[arow + fm * 16][ak0];
      af[fm] = pack16(*(const v8bf*)p, *(const v8bf*)(p + 16));
    }
#pragma unroll
    for (int fn = 0; fn < 4; ++fn) {
      const __bf16* p = &sB[bcol + fn * 16][bk0];
      bfr[fn] = pack16(*(const v8bf*)p, *(const v8bf*)(p + 16));
    }
#pragma unroll
    for (int fm = 0; fm < 2; ++fm)
#pragma unroll
      for (int fn = 0; fn < 4; ++fn)
        acc[fm][fn] = __builtin_amdgcn_wmma_f32_16x16x32_bf16(
            false, af[fm], false, bfr[fn], (short)0, acc[fm][fn], false, false);
    __syncthreads();
  }

  // ---- epilogue: C frag lane holds n = lane&15, m = (lane<16?0:8)+r ----
  const int nl = lane & 15;
  const int mb = (lane < 16) ? 0 : 8;
#pragma unroll
  for (int fm = 0; fm < 2; ++fm)
#pragma unroll
    for (int fn = 0; fn < 4; ++fn) {
      int gn = n0 + wn * 64 + fn * 16 + nl;
      if (gn >= N) continue;
      float bv = HASBIAS ? bias[gn] : 0.f;
#pragma unroll
      for (int r = 0; r < 8; ++r) {
        int gm = m0 + wm * 32 + fm * 16 + mb + r;
        if (gm < M) {
          float v = acc[fm][fn][r] + bv;
          if (RELU) v = fmaxf(v, 0.f);
          C[(size_t)gm * N + gn] = v;
        }
      }
    }
}

// ---------------------------------------------------------------------------
// small kernels
// ---------------------------------------------------------------------------
__global__ void zero_kernel(float* p, int n) {
  int i = blockIdx.x * 256 + threadIdx.x;
  if (i < n) p[i] = 0.f;
}

// concat(h2, time_emb[t]) -> (B, 1024)
__global__ __launch_bounds__(256) void build_concat(
    const float* __restrict__ h2, const float* __restrict__ tet,
    const int* __restrict__ tidx, float* __restrict__ out)
{
  int b = blockIdx.x;
  int t = tidx[b];
  for (int i = threadIdx.x; i < H_SZ; i += blockDim.x) {
    out[(size_t)b * (2 * H_SZ) + i]         = h2[(size_t)b * H_SZ + i];
    out[(size_t)b * (2 * H_SZ) + H_SZ + i]  = tet[(size_t)t * H_SZ + i];
  }
}

// mu = hidden@Wmu + bmu ; logvar = hidden@Wlv + blv   (N=50: plain VALU dot)
__global__ __launch_bounds__(64) void mulv_kernel(
    const float* __restrict__ hidden,
    const float* __restrict__ Wmu, const float* __restrict__ bmu,
    const float* __restrict__ Wlv, const float* __restrict__ blv,
    float* __restrict__ mu, float* __restrict__ lv)
{
  __shared__ float sh[H_SZ];
  int b = blockIdx.x;
  for (int i = threadIdx.x; i < H_SZ; i += 64) sh[i] = hidden[(size_t)b * H_SZ + i];
  __syncthreads();
  int j = threadIdx.x;
  if (j < K_TOP) {
    float s0 = bmu[j], s1 = blv[j];
    for (int h = 0; h < H_SZ; ++h) {
      float x = sh[h];
      s0 = fmaf(x, Wmu[h * K_TOP + j], s0);
      s1 = fmaf(x, Wlv[h * K_TOP + j], s1);
    }
    mu[b * K_TOP + j] = s0;
    lv[b * K_TOP + j] = s1;
  }
}

// theta = softmax(mu) rowwise; accumulate KL into acc[0]
__global__ __launch_bounds__(64) void softmax_theta_kl(
    const float* __restrict__ mu, const float* __restrict__ lv,
    float* __restrict__ theta, float* __restrict__ acc)
{
  __shared__ float buf[64];
  int b = blockIdx.x, j = threadIdx.x;
  float muv = (j < K_TOP) ? mu[b * K_TOP + j] : -3.402823466e38f;
  buf[j] = muv; __syncthreads();
#pragma unroll
  for (int o = 32; o > 0; o >>= 1) { if (j < o) buf[j] = fmaxf(buf[j], buf[j + o]); __syncthreads(); }
  float mx = buf[0]; __syncthreads();
  float e = (j < K_TOP) ? expf(muv - mx) : 0.f;
  buf[j] = e; __syncthreads();
#pragma unroll
  for (int o = 32; o > 0; o >>= 1) { if (j < o) buf[j] += buf[j + o]; __syncthreads(); }
  float s = buf[0]; __syncthreads();
  if (j < K_TOP) theta[b * K_TOP + j] = e / s;
  float kl = 0.f;
  if (j < K_TOP) {
    float l = lv[b * K_TOP + j];
    kl = -0.5f * (1.f + l - muv * muv - expf(l));
  }
  buf[j] = kl; __syncthreads();
#pragma unroll
  for (int o = 32; o > 0; o >>= 1) { if (j < o) buf[j] += buf[j + o]; __syncthreads(); }
  if (j == 0) atomicAdd(&acc[0], buf[0]);
}

// per-row (t,k) online max + sum(exp) over V for the logits matrix
__global__ __launch_bounds__(256) void rowstats_kernel(
    const float* __restrict__ logits, float* __restrict__ rmax,
    float* __restrict__ rinv, int ncol)
{
  __shared__ float sm[256], ss[256];
  int r = blockIdx.x, tid = threadIdx.x;
  const float* row = logits + (size_t)r * ncol;
  float m = -3.402823466e38f, s = 0.f;
  for (int i = tid; i < ncol; i += 256) {
    float x = row[i];
    if (x > m) { s = s * expf(m - x) + 1.f; m = x; }
    else       { s += expf(x - m); }
  }
  sm[tid] = m; ss[tid] = s; __syncthreads();
#pragma unroll
  for (int o = 128; o > 0; o >>= 1) {
    if (tid < o) {
      float m2 = sm[tid + o], s2 = ss[tid + o];
      float M  = fmaxf(sm[tid], m2);
      ss[tid] = ss[tid] * expf(sm[tid] - M) + s2 * expf(m2 - M);
      sm[tid] = M;
    }
    __syncthreads();
  }
  if (tid == 0) { rmax[r] = sm[0]; rinv[r] = 1.f / ss[0]; }
}

// betaT[t][v][k] = exp(logits[r,v]-max_r)*inv_r   (r = t*50+k), 32x32 LDS tiles
__global__ __launch_bounds__(256) void beta_transpose(
    const float* __restrict__ logits, const float* __restrict__ rmax,
    const float* __restrict__ rinv, float* __restrict__ betaT, int R, int V)
{
  __shared__ float t[32][33];
  int r0 = blockIdx.y * 32, v0 = blockIdx.x * 32;
  int tx = threadIdx.x & 31, ty = threadIdx.x >> 5;   // 32 x 8
#pragma unroll
  for (int i = 0; i < 4; ++i) {
    int r = r0 + ty + i * 8, v = v0 + tx;
    t[ty + i * 8][tx] = (r < R && v < V)
        ? expf(logits[(size_t)r * V + v] - rmax[r]) * rinv[r] : 0.f;
  }
  __syncthreads();
#pragma unroll
  for (int i = 0; i < 4; ++i) {
    int r = r0 + tx, v = v0 + ty + i * 8;
    if (r < R && v < V) {
      int tt = r / K_TOP, k = r % K_TOP;
      betaT[((size_t)tt * V + v) * K_TOP + k] = t[tx][ty + i * 8];
    }
  }
}

// fused word_dist + recon partials: ps[b] += sum(bow), ps[B+b] += sum(bow*log(wd+eps))
// TDM path: DMA 256x50-f32 slabs of betaT[t] into LDS (TENSOR_LOAD_TO_LDS, one
// per tile from wave 0, TENSORcnt-synced), then each thread dots its row out of
// LDS (conflict-free: 50-dword row stride, gcd(50,64)=2 -> collisions only 32
// lanes apart). bow stays a directly coalesced global stream.
__global__ __launch_bounds__(256) void recon_partial(
    const float* __restrict__ bow, const float* __restrict__ theta,
    const int* __restrict__ tidx, const float* __restrict__ betaT,
    float* __restrict__ ps)
{
  __shared__ float sth[K_TOP];
  __shared__ float b0[256], b1[256];
#if HAVE_TDM
  __shared__ float tile[256 * K_TOP];   // 51200 B slab
#endif
  int b = blockIdx.y;
  if (threadIdx.x < K_TOP) sth[threadIdx.x] = theta[b * K_TOP + threadIdx.x];
  __syncthreads();
  int t = tidx[b];
  const float* bt = betaT + (size_t)t * V_SZ * K_TOP;
  const float* br = bow + (size_t)b * V_SZ;
  float s0 = 0.f, s1 = 0.f;
  const int chunk = blockIdx.x * 2048;

#if HAVE_TDM
  for (int j = 0; j < 8; ++j) {
    int base = chunk + j * 256;
    int rows = V_SZ - base;
    if (rows > 256) rows = 256;
    __syncthreads();                       // previous slab fully consumed
    if (threadIdx.x < 32) {                // wave 0 drives the DMA
      if (rows > 0)
        tdm_1d_load((unsigned)(size_t)&tile[0],
                    (unsigned long long)(size_t)(bt + (size_t)base * K_TOP),
                    rows * K_TOP);
      wait_tensorcnt0();
    }
    __syncthreads();
    int v = base + threadIdx.x;
    if (v < V_SZ) {
      const float* bp = &tile[threadIdx.x * K_TOP];
      float wd = 0.f;
#pragma unroll
      for (int k = 0; k < K_TOP; ++k) wd = fmaf(sth[k], bp[k], wd);
      float bv = br[v];
      s0 += bv;
      s1 += bv * logf(wd + EPSF);
    }
  }
#else
  int v = chunk + threadIdx.x;
#pragma unroll
  for (int i = 0; i < 8; ++i, v += 256) {
    if (v < V_SZ) {
      const float* bp = bt + (size_t)v * K_TOP;
      __builtin_prefetch(br + v, 0, 1);
      float wd = 0.f;
#pragma unroll
      for (int k = 0; k < K_TOP; ++k) wd = fmaf(sth[k], bp[k], wd);
      float bv = br[v];
      s0 += bv;
      s1 += bv * logf(wd + EPSF);
    }
  }
#endif

  b0[threadIdx.x] = s0; b1[threadIdx.x] = s1; __syncthreads();
#pragma unroll
  for (int o = 128; o > 0; o >>= 1) {
    if (threadIdx.x < o) { b0[threadIdx.x] += b0[threadIdx.x + o]; b1[threadIdx.x] += b1[threadIdx.x + o]; }
    __syncthreads();
  }
  if (threadIdx.x == 0) {
    atomicAdd(&ps[b], b0[0]);
    atomicAdd(&ps[B_SZ + b], b1[0]);
  }
}

// recon = mean_b [ -dot_b / (bowsum_b + eps) ] -> acc[2]
__global__ __launch_bounds__(256) void recon_final(const float* __restrict__ ps, float* __restrict__ acc) {
  __shared__ float buf[256];
  float s = 0.f;
  for (int b = threadIdx.x; b < B_SZ; b += 256)
    s += -ps[B_SZ + b] / (ps[b] + EPSF);
  s = block_reduce_sum256(s, buf);
  if (threadIdx.x == 0) acc[2] = s / (float)B_SZ;
}

// evolution loss: sum of (te[i+KD]-te[i])^2 over 9*K*D -> acc[1]
__global__ __launch_bounds__(256) void evo_kernel(
    const float* __restrict__ te, float* __restrict__ acc, int n, int off)
{
  __shared__ float buf[256];
  float s = 0.f;
  for (int i = blockIdx.x * blockDim.x + threadIdx.x; i < n; i += gridDim.x * blockDim.x) {
    float d = te[i + off] - te[i];
    s = fmaf(d, d, s);
  }
  s = block_reduce_sum256(s, buf);
  if (threadIdx.x == 0) atomicAdd(&acc[1], s);
}

__global__ void finalize_kernel(const float* __restrict__ acc, float* __restrict__ out) {
  if (threadIdx.x == 0) {
    float kl    = acc[0] / (float)B_SZ;
    float evo   = acc[1] / (float)((T_SZ - 1) * K_TOP * D_SZ);
    float recon = acc[2];
    out[0] = recon + 0.5f * kl + 0.1f * evo;
    out[1] = recon;
    out[2] = kl;
    out[3] = evo;
  }
}

// ---------------------------------------------------------------------------
extern "C" void kernel_launch(void* const* d_in, const int* in_sizes, int n_in,
                              void* d_out, int out_size, void* d_ws, size_t ws_size,
                              hipStream_t stream) {
  const float* doc  = (const float*)d_in[0];
  const float* bow  = (const float*)d_in[1];
  const int*   tidx = (const int*)  d_in[2];
  const float* W1   = (const float*)d_in[3];
  const float* b1   = (const float*)d_in[4];
  const float* W2   = (const float*)d_in[5];
  const float* b2   = (const float*)d_in[6];
  const float* tet  = (const float*)d_in[7];
  const float* Wf   = (const float*)d_in[8];
  const float* bfv  = (const float*)d_in[9];
  const float* Wmu  = (const float*)d_in[10];
  const float* bmu  = (const float*)d_in[11];
  const float* Wlv  = (const float*)d_in[12];
  const float* blv  = (const float*)d_in[13];
  const float* wemb = (const float*)d_in[14];
  const float* temb = (const float*)d_in[15];   // (T,K,D) == (500, 1024) flat
  float* out = (float*)d_out;

  char* w = (char*)d_ws;
  size_t off = 0;
  auto alloc = [&](size_t nbytes) -> float* {
    float* p = (float*)(w + off);
    off = (off + nbytes + 255) & ~(size_t)255;
    return p;
  };
  float* h1     = alloc((size_t)B_SZ * H_SZ * 4);
  float* h2     = alloc((size_t)B_SZ * H_SZ * 4);
  float* cc     = alloc((size_t)B_SZ * 2 * H_SZ * 4);
  float* hid    = alloc((size_t)B_SZ * H_SZ * 4);
  float* mu     = alloc((size_t)B_SZ * K_TOP * 4);
  float* lv     = alloc((size_t)B_SZ * K_TOP * 4);
  float* logits = alloc((size_t)R_SZ * V_SZ * 4);
  float* rmax   = alloc((size_t)R_SZ * 4);
  float* rinv   = alloc((size_t)R_SZ * 4);
  float* betaT  = alloc((size_t)T_SZ * V_SZ * K_TOP * 4);
  float* ps     = alloc((size_t)(2 * B_SZ + 16) * 4);
  float* acc    = ps + 2 * B_SZ;
  float* theta  = out + 4;

  // accumulators must be zero every call (deterministic, graph-replayable)
  zero_kernel<<<(2 * B_SZ + 16 + 255) / 256, 256, 0, stream>>>(ps, 2 * B_SZ + 16);

  dim3 blk(256);
  // encoder: h1 = relu(doc @ W1 + b1)
  gemm_bf16_wmma<false, true, true><<<dim3(H_SZ / 128, B_SZ / 128), blk, 0, stream>>>(
      doc, W1, b1, h1, B_SZ, H_SZ, D_SZ);
  // h2 = relu(h1 @ W2 + b2)
  gemm_bf16_wmma<false, true, true><<<dim3(H_SZ / 128, B_SZ / 128), blk, 0, stream>>>(
      h1, W2, b2, h2, B_SZ, H_SZ, H_SZ);
  // concat(h2, time_emb[t])
  build_concat<<<B_SZ, 256, 0, stream>>>(h2, tet, tidx, cc);
  // hidden = relu(cc @ Wf + bf)
  gemm_bf16_wmma<false, true, true><<<dim3(H_SZ / 128, B_SZ / 128), blk, 0, stream>>>(
      cc, Wf, bfv, hid, B_SZ, H_SZ, 2 * H_SZ);
  // mu / logvar
  mulv_kernel<<<B_SZ, 64, 0, stream>>>(hid, Wmu, bmu, Wlv, blv, mu, lv);
  // theta + KL
  softmax_theta_kl<<<B_SZ, 64, 0, stream>>>(mu, lv, theta, acc);

  // decoder logits: (500 x 50000) = topic_emb(500x1024) * word_emb(50000x1024)^T
  gemm_bf16_wmma<true, false, false><<<dim3((V_SZ + 127) / 128, (R_SZ + 127) / 128), blk, 0, stream>>>(
      temb, wemb, nullptr, logits, R_SZ, V_SZ, D_SZ);
  // softmax stats per (t,k) row
  rowstats_kernel<<<R_SZ, 256, 0, stream>>>(logits, rmax, rinv, V_SZ);
  // normalized betas, transposed to (T,V,K) for the contiguous dot50
  beta_transpose<<<dim3((V_SZ + 31) / 32, (R_SZ + 31) / 32), 256, 0, stream>>>(
      logits, rmax, rinv, betaT, R_SZ, V_SZ);
  // fused word_dist + recon partials (single pass over bow, TDM-staged betas)
  recon_partial<<<dim3((V_SZ + 2047) / 2048, B_SZ), 256, 0, stream>>>(
      bow, theta, tidx, betaT, ps);
  recon_final<<<1, 256, 0, stream>>>(ps, acc);
  // evolution loss
  evo_kernel<<<256, 256, 0, stream>>>(temb, acc, (T_SZ - 1) * K_TOP * D_SZ, K_TOP * D_SZ);
  // combine
  finalize_kernel<<<1, 32, 0, stream>>>(acc, out);
}
